// LinkPredictor_65000035058080
// MI455X (gfx1250) — matrix operand order repaired and checked by reading, not compiled, not run on hardware
//
#include <hip/hip_runtime.h>

typedef __attribute__((ext_vector_type(16))) __bf16 v16bf;
typedef __attribute__((ext_vector_type(8)))  __bf16 v8bf;
typedef __attribute__((ext_vector_type(8)))  float  v8f;
typedef __attribute__((ext_vector_type(4)))  float  v4f;
typedef __attribute__((ext_vector_type(4)))  unsigned int v4u;

#define N_EDGES_C 1000000
#define F_DIM_C   128

// Packed bf16 weight-fragment layout in d_ws (element offsets, bf16):
//   each 32x16 (KxN) tile -> 512 bf16: lane l (0..31) holds 16 K-values for
//   column N = 16*nt + (l&15); lanes 0-15 cover K = 32*kt+{0..7,16..23}, lanes 16-31
//   cover K = 32*kt+{8..15,24..31} (ISA 7.12.2 B-matrix striping).
#define OFF_W1 0        // 256x64 -> 8 ktiles x 4 ntiles -> 16384
#define OFF_A1 16384
#define OFF_W2 32768    // 64x32 -> 2x2 -> 2048
#define OFF_A2 34816
#define OFF_W3 36864    // 32x16 -> 1x1 -> 512
#define OFF_A3 37376
#define WPACK_TOTAL 37888

__device__ __forceinline__ void pack_one(const float* M, int N, int NT,
                                         __bf16* dst, int tid) {
  int i  = tid & 15;           // position within lane's 16 values
  int l  = (tid >> 4) & 31;    // lane
  int f  = tid >> 9;           // fragment id
  int kt = f / NT, nt = f % NT;
  int k  = 32 * kt + ((l >= 16) ? 8 : 0) + ((i < 8) ? i : i + 8);
  int n  = 16 * nt + (l & 15);
  dst[tid] = (__bf16)M[k * N + n];
}

__global__ void pack_weights_kernel(const float* __restrict__ W1, const float* __restrict__ A1,
                                    const float* __restrict__ W2, const float* __restrict__ A2,
                                    const float* __restrict__ W3, const float* __restrict__ A3,
                                    __bf16* __restrict__ ws) {
  int tid = blockIdx.x * blockDim.x + threadIdx.x;
  if      (tid < 16384) pack_one(W1, 64, 4, ws + OFF_W1, tid);
  else if (tid < 32768) pack_one(A1, 64, 4, ws + OFF_A1, tid - 16384);
  else if (tid < 34816) pack_one(W2, 32, 2, ws + OFF_W2, tid - 32768);
  else if (tid < 36864) pack_one(A2, 32, 2, ws + OFF_A2, tid - 34816);
  else if (tid < 37376) pack_one(W3, 16, 1, ws + OFF_W3, tid - 36864);
  else if (tid < 37888) pack_one(A3, 16, 1, ws + OFF_A3, tid - 37376);
}

__device__ __forceinline__ v8f wmma_bf16(v16bf a, v16bf b, v8f c) {
  return __builtin_amdgcn_wmma_f32_16x16x32_bf16(false, a, false, b,
                                                 (short)0, c, false, false);
}

__device__ __forceinline__ v16bf frag_from_lds(const __bf16* base, int lane) {
  union { v16bf v; v8bf h[2]; } u;
  const v8bf* q = (const v8bf*)(base + lane * 16);
  u.h[0] = q[0];
  u.h[1] = q[1];
  return u.v;
}

// Build 16-bit A-fragment per ISA 7.12.2: lane row holds K = c0+{0..7} and
// c0+{16..23} (c0 already includes the +8 shift for hi lanes).
__device__ __forceinline__ v16bf gather_frag(const float* row, int c0) {
  const v4f* r = (const v4f*)row;
  v4f f0 = r[(c0 >> 2) + 0];
  v4f f1 = r[(c0 >> 2) + 1];
  v4f f2 = r[(c0 >> 2) + 4];
  v4f f3 = r[(c0 >> 2) + 5];
  union { v16bf v; __bf16 e[16]; } u;
#pragma unroll
  for (int i = 0; i < 4; ++i) {
    u.e[i]      = (__bf16)f0[i];
    u.e[4 + i]  = (__bf16)f1[i];
    u.e[8 + i]  = (__bf16)f2[i];
    u.e[12 + i] = (__bf16)f3[i];
  }
  return u.v;
}

// Each wave handles TWO 16-edge tiles (M=32) so every B-fragment LDS load
// feeds two WMMAs -> halves LDS bandwidth per edge. launch_bounds(256,1)
// lifts the VGPR cap so the 16 accumulators stay register-resident.
__global__ void __launch_bounds__(256, 1)
edge_mlp_kernel(const float* __restrict__ x,
                const long long* __restrict__ ei,
                const float* __restrict__ b1,
                const float* __restrict__ b2,
                const float* __restrict__ b3,
                const float* __restrict__ Wf,
                const float* __restrict__ bf,
                const __bf16* __restrict__ wpack,
                float* __restrict__ out,
                int n_pairs) {
  extern __shared__ char smem_raw[];
  __bf16* sw = (__bf16*)smem_raw;                 // [WPACK_TOTAL] weights

  // Cooperative stage of packed weights into LDS (16B copies).
  {
    const v4u* src = (const v4u*)wpack;
    v4u* dst = (v4u*)sw;
    for (int i = threadIdx.x; i < (WPACK_TOTAL * 2 / 16); i += blockDim.x)
      dst[i] = src[i];
  }
  __syncthreads();

  const int lane   = threadIdx.x & 31;
  const int warp   = threadIdx.x >> 5;
  const int laneLo = lane & 15;
  const bool hi    = lane >= 16;
  const int hsh    = hi ? 8 : 0;

  __bf16* hbuf = sw + WPACK_TOTAL + warp * 2048;  // per-wave 2x(16x64) bf16 scratch

  const int pair = blockIdx.x * (blockDim.x >> 5) + warp;
  if (pair >= n_pairs) return;

  const int ebase = pair * 32;
  const int e0 = ebase + laneLo;
  const int e1 = ebase + 16 + laneLo;
  const float* srow0 = x + ei[e0] * F_DIM_C;
  const float* drow0 = x + ei[N_EDGES_C + e0] * F_DIM_C;
  const float* srow1 = x + ei[e1] * F_DIM_C;
  const float* drow1 = x + ei[N_EDGES_C + e1] * F_DIM_C;

  // dst rows are consumed in the second half of the K loop: prefetch now.
  __builtin_prefetch(drow0 + hsh, 0, 1);
  __builtin_prefetch(drow1 + hsh, 0, 1);

  // ---------------- Layer 1: [32x256] @ {W1,A1}[256x64] -------------------
  v8f accW[2][4], accA[2][4];
#pragma unroll
  for (int s = 0; s < 2; ++s)
#pragma unroll
    for (int n = 0; n < 4; ++n) {
      v8f z = {0.f, 0.f, 0.f, 0.f, 0.f, 0.f, 0.f, 0.f};
      accW[s][n] = z; accA[s][n] = z;
    }

#pragma unroll
  for (int k = 0; k < 8; ++k) {
    const float* row0 = (k < 4) ? srow0 : drow0;  // concat: first half src, second dst
    const float* row1 = (k < 4) ? srow1 : drow1;
    int c0 = 32 * (k & 3) + hsh;
    v16bf a0 = gather_frag(row0, c0);
    v16bf a1 = gather_frag(row1, c0);
#pragma unroll
    for (int n = 0; n < 4; ++n) {
      v16bf bw = frag_from_lds(sw + OFF_W1 + (k * 4 + n) * 512, lane);
      accW[0][n] = wmma_bf16(a0, bw, accW[0][n]);
      accW[1][n] = wmma_bf16(a1, bw, accW[1][n]);
      v16bf ba = frag_from_lds(sw + OFF_A1 + (k * 4 + n) * 512, lane);
      accA[0][n] = wmma_bf16(a0, ba, accA[0][n]);
      accA[1][n] = wmma_bf16(a1, ba, accA[1][n]);
    }
  }

  // Epilogue 1: relu(relu(hW+b1) + hA) -> hbuf[2][16x64] bf16 (D->A re-layout)
#pragma unroll
  for (int s = 0; s < 2; ++s)
#pragma unroll
    for (int n = 0; n < 4; ++n) {
      float bv = b1[16 * n + laneLo];
#pragma unroll
      for (int j = 0; j < 8; ++j) {
        float t = fmaxf(accW[s][n][j] + bv, 0.0f) + accA[s][n][j];
        t = fmaxf(t, 0.0f);
        hbuf[s * 1024 + (j + hsh) * 64 + 16 * n + laneLo] = (__bf16)t;
      }
    }
  __builtin_amdgcn_fence(__ATOMIC_ACQ_REL, "wavefront");
  __builtin_amdgcn_wave_barrier();

  // ---------------- Layer 2: [32x64] @ {W2,A2}[64x32] ---------------------
  v16bf a2[2][2];
#pragma unroll
  for (int s = 0; s < 2; ++s)
#pragma unroll
    for (int kk = 0; kk < 2; ++kk) {
      int base = 32 * kk + hsh;
      union { v16bf v; v8bf h[2]; } u;
      u.h[0] = *(const v8bf*)(hbuf + s * 1024 + laneLo * 64 + base);
      u.h[1] = *(const v8bf*)(hbuf + s * 1024 + laneLo * 64 + base + 16);
      a2[s][kk] = u.v;
    }
  v8f acc2W[2][2], acc2A[2][2];
#pragma unroll
  for (int s = 0; s < 2; ++s)
#pragma unroll
    for (int n = 0; n < 2; ++n) {
      v8f z = {0.f, 0.f, 0.f, 0.f, 0.f, 0.f, 0.f, 0.f};
      acc2W[s][n] = z; acc2A[s][n] = z;
    }
#pragma unroll
  for (int kk = 0; kk < 2; ++kk)
#pragma unroll
    for (int n = 0; n < 2; ++n) {
      v16bf bw = frag_from_lds(sw + OFF_W2 + (kk * 2 + n) * 512, lane);
      acc2W[0][n] = wmma_bf16(a2[0][kk], bw, acc2W[0][n]);
      acc2W[1][n] = wmma_bf16(a2[1][kk], bw, acc2W[1][n]);
      v16bf ba = frag_from_lds(sw + OFF_A2 + (kk * 2 + n) * 512, lane);
      acc2A[0][n] = wmma_bf16(a2[0][kk], ba, acc2A[0][n]);
      acc2A[1][n] = wmma_bf16(a2[1][kk], ba, acc2A[1][n]);
    }
  __builtin_amdgcn_wave_barrier();   // all reads of hbuf done before overwrite

  // Epilogue 2 -> hbuf[2][16x32] bf16
#pragma unroll
  for (int s = 0; s < 2; ++s)
#pragma unroll
    for (int n = 0; n < 2; ++n) {
      float bv = b2[16 * n + laneLo];
#pragma unroll
      for (int j = 0; j < 8; ++j) {
        float t = fmaxf(acc2W[s][n][j] + bv, 0.0f) + acc2A[s][n][j];
        t = fmaxf(t, 0.0f);
        hbuf[s * 1024 + (j + hsh) * 32 + 16 * n + laneLo] = (__bf16)t;
      }
    }
  __builtin_amdgcn_fence(__ATOMIC_ACQ_REL, "wavefront");
  __builtin_amdgcn_wave_barrier();

  // ---------------- Layer 3: [32x32] @ {W3,A3}[32x16] ---------------------
  v16bf bw3 = frag_from_lds(sw + OFF_W3, lane);
  v16bf ba3 = frag_from_lds(sw + OFF_A3, lane);
  v8f acc3W[2], acc3A[2];
#pragma unroll
  for (int s = 0; s < 2; ++s) {
    int base = hsh;
    union { v16bf v; v8bf h[2]; } u;
    u.h[0] = *(const v8bf*)(hbuf + s * 1024 + laneLo * 32 + base);
    u.h[1] = *(const v8bf*)(hbuf + s * 1024 + laneLo * 32 + base + 16);
    v8f z = {0.f, 0.f, 0.f, 0.f, 0.f, 0.f, 0.f, 0.f};
    acc3W[s] = wmma_bf16(u.v, bw3, z);
    acc3A[s] = wmma_bf16(u.v, ba3, z);
  }

  // ---------------- Final: sigmoid(h3 @ Wf + bf) --------------------------
  const float b3v = b3[laneLo];
  const float wfv = Wf[laneLo];
  const float bfv = bf[0];
#pragma unroll
  for (int s = 0; s < 2; ++s)
#pragma unroll
    for (int j = 0; j < 8; ++j) {
      float t = fmaxf(acc3W[s][j] + b3v, 0.0f) + acc3A[s][j];
      t = fmaxf(t, 0.0f);
      float p = t * wfv;               // partial over column N = laneLo
      p += __shfl_xor(p, 1);           // butterfly within 16-lane half
      p += __shfl_xor(p, 2);
      p += __shfl_xor(p, 4);
      p += __shfl_xor(p, 8);
      if (laneLo == j) {
        float zv = p + bfv;
        out[ebase + s * 16 + j + hsh] = 1.0f / (1.0f + __expf(-zv));
      }
    }
}

extern "C" void kernel_launch(void* const* d_in, const int* in_sizes, int n_in,
                              void* d_out, int out_size, void* d_ws, size_t ws_size,
                              hipStream_t stream) {
  const float*     x  = (const float*)d_in[0];
  const long long* ei = (const long long*)d_in[1];
  const float* W1 = (const float*)d_in[2];
  const float* b1 = (const float*)d_in[3];
  const float* A1 = (const float*)d_in[4];
  const float* W2 = (const float*)d_in[5];
  const float* b2 = (const float*)d_in[6];
  const float* A2 = (const float*)d_in[7];
  const float* W3 = (const float*)d_in[8];
  const float* b3 = (const float*)d_in[9];
  const float* A3 = (const float*)d_in[10];
  const float* Wf = (const float*)d_in[11];
  const float* bf = (const float*)d_in[12];
  __bf16* wpack = (__bf16*)d_ws;
  float*  out   = (float*)d_out;

  pack_weights_kernel<<<(WPACK_TOTAL + 255) / 256, 256, 0, stream>>>(
      W1, A1, W2, A2, W3, A3, wpack);

  const int n_pairs = N_EDGES_C / 32;                 // 31250 (two tiles per wave)
  const int waves_per_block = 8;                      // 256 threads, wave32
  const int blocks = (n_pairs + waves_per_block - 1) / waves_per_block;
  const size_t smem = (size_t)(WPACK_TOTAL * 2 + waves_per_block * 2048 * 2); // 108544 B

  static bool attr_set = []() {
    hipFuncSetAttribute((const void*)edge_mlp_kernel,
                        hipFuncAttributeMaxDynamicSharedMemorySize, 131072);
    return true;
  }();
  (void)attr_set;

  edge_mlp_kernel<<<blocks, 256, smem, stream>>>(
      x, ei, b1, b2, b3, Wf, bf, wpack, out, n_pairs);
}
